// MHA_71193377898510
// MI455X (gfx1250) — matrix operand order repaired and checked by reading, not compiled, not run on hardware
//
#include <hip/hip_runtime.h>
#include <hip/hip_bf16.h>
#include <math.h>

#define BB 2
#define SS 2048
#define DD 1024
#define HH 16
#define KDIM 64

typedef __bf16 bf16;
typedef __attribute__((ext_vector_type(16))) __bf16 v16bf;
typedef __attribute__((ext_vector_type(8)))  __bf16 v8bf;
typedef __attribute__((ext_vector_type(4)))  __bf16 v4bf;
typedef __attribute__((ext_vector_type(8)))  float  v8f;
typedef __attribute__((ext_vector_type(4)))  int    i32x4;

#define AS1 __attribute__((address_space(1)))
#define AS3 __attribute__((address_space(3)))
typedef AS1 i32x4* g_v4i_ptr;   // pointer to int4 in global AS
typedef AS3 i32x4* l_v4i_ptr;   // pointer to int4 in LDS AS

#if __has_builtin(__builtin_amdgcn_global_load_async_to_lds_b128)
#define HAS_ASYNC 1
#else
#define HAS_ASYNC 0
#endif

// Stage one 16B chunk global->LDS. Async path uses the CDNA5
// GLOBAL_LOAD_ASYNC_TO_LDS_B128 (ASYNCcnt); fallback is a b128 round trip.
__device__ __forceinline__ void stage16(bf16* l, const bf16* g) {
#if HAS_ASYNC
    __builtin_amdgcn_global_load_async_to_lds_b128(
        (g_v4i_ptr)(unsigned long long)g,
        (l_v4i_ptr)(unsigned int)(unsigned long long)l, 0, 0);
#else
    *(v8bf*)l = *(const v8bf*)g;
#endif
}

__device__ __forceinline__ void async_wait0() {
#if HAS_ASYNC
#if __has_builtin(__builtin_amdgcn_s_wait_asynccnt)
    __builtin_amdgcn_s_wait_asynccnt(0);
#else
    asm volatile("s_wait_asynccnt 0" ::: "memory");
#endif
#endif
}

// ---------------------------------------------------------------------------
// Fragment loaders matching CDNA5 ISA 7.12.2 VGPR layouts (wave32).
// A (16x32 bf16): lane L holds row M=L&15; K(e) = 16*(e>>3) + 8*(L>>4) + (e&7)
//   -> two contiguous 16B reads at K offsets 8*half and 16+8*half.
// B (32x16 bf16): lane L holds col N=L&15; K(e) = 16*(e>>3) + 8*(L>>4) + (e&7)
//   -> two contiguous 16B reads at K offsets 8*half and 16+8*half (n-major tile).
// ---------------------------------------------------------------------------
__device__ __forceinline__ v16bf load_a_frag(const bf16* base, int ldk) {
    const int lane = threadIdx.x & 31;
    const int m = lane & 15, half = lane >> 4;
    const bf16* row = base + m * ldk + 8 * half;
    union { v16bf v; v8bf h[2]; } u;
    u.h[0] = *(const v8bf*)(row);
    u.h[1] = *(const v8bf*)(row + 16);
    return u.v;
}

__device__ __forceinline__ v16bf load_b_frag(const bf16* bt, int ldk) {
    const int lane = threadIdx.x & 31;
    const int n = lane & 15, half = lane >> 4;
    const bf16* p = bt + n * ldk + 8 * half;
    union { v16bf v; v8bf h[2]; } u;
    u.h[0] = *(const v8bf*)(p);
    u.h[1] = *(const v8bf*)(p + 16);
    return u.v;
}

__device__ __forceinline__ v8f wmma_bf16(v16bf a, v16bf b, v8f c) {
    return __builtin_amdgcn_wmma_f32_16x16x32_bf16(false, a, false, b,
                                                   (short)0, c, false, false);
}

// ---------------------------------------------------------------------------
// fp32 -> bf16 convert (x), and fused transpose+convert (weights).
// ---------------------------------------------------------------------------
__global__ void cvt_kernel(const float4* __restrict__ in, v4bf* __restrict__ out, int n4) {
    int i = blockIdx.x * blockDim.x + threadIdx.x;
    if (i < n4) {
        float4 f = in[i];
        v4bf o;
        o[0] = (bf16)f.x; o[1] = (bf16)f.y; o[2] = (bf16)f.z; o[3] = (bf16)f.w;
        out[i] = o;
    }
}

// in: fp32 [K][N] row-major; out: bf16 [N][K] row-major (i.e. W^T).
__global__ __launch_bounds__(256)
void cvt_t_kernel(const float* __restrict__ in, bf16* __restrict__ out, int K, int N) {
    __shared__ float tile[32][33];
    const int k0 = blockIdx.x * 32, n0 = blockIdx.y * 32;
    const int t = threadIdx.x;
    {
        int k = t >> 3, n4 = (t & 7) * 4;
        float4 f = *(const float4*)(in + (size_t)(k0 + k) * N + n0 + n4);
        tile[k][n4 + 0] = f.x; tile[k][n4 + 1] = f.y;
        tile[k][n4 + 2] = f.z; tile[k][n4 + 3] = f.w;
    }
    __syncthreads();
    {
        int n = t >> 3, k4 = (t & 7) * 4;
        v4bf o;
        o[0] = (bf16)tile[k4 + 0][n]; o[1] = (bf16)tile[k4 + 1][n];
        o[2] = (bf16)tile[k4 + 2][n]; o[3] = (bf16)tile[k4 + 3][n];
        *(v4bf*)(out + (size_t)(n0 + n) * K + k0 + k4) = o;
    }
}

// ---------------------------------------------------------------------------
// Tiled bf16 GEMM: out = A[M,K] @ W + bias, with W given PRE-TRANSPOSED as
// Wt[N][K]. 256 threads = 8 waves; tile 256x64, K-step 32; each wave owns a
// 32x64 strip -> 8 WMMAs per K-step.
// mode 0: out bf16 [b][h][s][kd]   (Q, K projections)
// mode 1: out bf16 [b][h][kd][s]   (V projection, pre-transposed)
// mode 2: out f32 row-major [M][N] (output projection)
// ---------------------------------------------------------------------------
#define GBM 256
#define GBN 64
#define GBK 32
#define A_LD 40   // 32 + pad; 80B row stride keeps every frag read 16B aligned
#define W_LD 40

__global__ __launch_bounds__(256)
void gemm_kernel(const bf16* __restrict__ A, const bf16* __restrict__ Wt,
                 const float* __restrict__ bias, void* __restrict__ outp,
                 int M, int N, int K, int mode) {
    __shared__ __attribute__((aligned(16))) bf16 At[GBM * A_LD];
    __shared__ __attribute__((aligned(16))) bf16 Ws[GBN * W_LD];

    const int t    = threadIdx.x;
    const int wave = t >> 5;
    const int lane = t & 31;
    const int nl = lane & 15, half = lane >> 4;
    const int m0 = blockIdx.x * GBM;
    const int n0 = blockIdx.y * GBN;

    v8f acc[2][4] = {};

    for (int k0 = 0; k0 < K; k0 += GBK) {
        __syncthreads();
        // A tile 256x32 (1024 16B chunks, 4 per thread)
#pragma unroll
        for (int i = 0; i < 4; i++) {
            int c = t + i * 256;
            int r = c >> 2, off = (c & 3) * 8;
            stage16(&At[r * A_LD + off], A + (size_t)(m0 + r) * K + k0 + off);
        }
        // W^T tile 64x32 (256 chunks, 1 per thread)
        {
            int r = t >> 2, off = (t & 3) * 8;
            stage16(&Ws[r * W_LD + off], Wt + (size_t)(n0 + r) * K + k0 + off);
        }
        async_wait0();
        __syncthreads();

        v16bf af0 = load_a_frag(&At[(wave * 32)      * A_LD], A_LD);
        v16bf af1 = load_a_frag(&At[(wave * 32 + 16) * A_LD], A_LD);
#pragma unroll
        for (int ct = 0; ct < 4; ct++) {
            v16bf bfrag = load_b_frag(&Ws[(ct * 16) * W_LD], W_LD);
            acc[0][ct] = wmma_bf16(af0, bfrag, acc[0][ct]);
            acc[1][ct] = wmma_bf16(af1, bfrag, acc[1][ct]);
        }
    }

    // Epilogue: bias add + store per output mode.
#pragma unroll
    for (int at = 0; at < 2; at++) {
#pragma unroll
        for (int ct = 0; ct < 4; ct++) {
            int gcol = n0 + ct * 16 + nl;
            float bvv = bias[gcol];
#pragma unroll
            for (int r = 0; r < 8; r++) {
                int grow = m0 + wave * 32 + at * 16 + r + 8 * half;
                float val = acc[at][ct][r] + bvv;
                if (mode == 2) {
                    ((float*)outp)[(size_t)grow * N + gcol] = val;
                } else {
                    int b = grow >> 11, s = grow & (SS - 1);
                    int h = gcol >> 6, kd = gcol & (KDIM - 1);
                    bf16* o = (bf16*)outp;
                    if (mode == 0)
                        o[((size_t)(b * HH + h) * SS + s) * KDIM + kd] = (bf16)val;
                    else
                        o[((size_t)(b * HH + h) * KDIM + kd) * SS + s] = (bf16)val;
                }
            }
        }
    }
}

// ---------------------------------------------------------------------------
// Flash attention: 1 WG = (b, h, 64 query rows); 4 waves x 16 rows.
// Q[b][h][s][kd], K[b][h][s][kd], Vt[b][h][kd][s] all bf16.
// Writes attended bf16 as [b][s][h*KD+kd] (feeds output GEMM directly).
// ---------------------------------------------------------------------------
#define AT_LD 80  // 64 + pad (160B stride, 16B aligned everywhere)

__global__ __launch_bounds__(128)
void attn_kernel(const bf16* __restrict__ Q, const bf16* __restrict__ Kg,
                 const bf16* __restrict__ Vt, bf16* __restrict__ attw) {
    __shared__ __attribute__((aligned(16))) bf16 Kt[64 * AT_LD];
    __shared__ __attribute__((aligned(16))) bf16 Vs[64 * AT_LD];
    __shared__ __attribute__((aligned(16))) bf16 Ps[64 * AT_LD];  // 4 waves x 16 rows

    const int t    = threadIdx.x;
    const int wave = t >> 5;
    const int lane = t & 31;
    const int nl = lane & 15, half = lane >> 4;

    const int wg = blockIdx.x;
    const int qb = wg & 31;           // S/64 = 32 query blocks
    const int h  = (wg >> 5) & 15;
    const int b  = wg >> 9;
    const size_t bh = (size_t)(b * HH + h);

    // Q fragments for this wave's 16 rows stay in registers for the whole loop.
    const bf16* qbase = Q + (bh * SS + (size_t)qb * 64 + wave * 16) * KDIM;
    v16bf qf0 = load_a_frag(qbase,      KDIM);
    v16bf qf1 = load_a_frag(qbase + 32, KDIM);

    v8f oacc[4] = {};
    float mrow[8], lrow[8];
#pragma unroll
    for (int r = 0; r < 8; r++) { mrow[r] = -INFINITY; lrow[r] = 0.0f; }
    const float scale = 0.125f;  // 1/sqrt(64)

    bf16* Pw = &Ps[(wave * 16) * AT_LD];

    for (int kb = 0; kb < SS / 64; kb++) {
        __syncthreads();
        { // Stage K block [64 keys][64 kd] and V^T block [64 kd][64 keys]
          // 512 16B chunks each; 4 chunks per tile per thread (async).
            const bf16* gk = Kg + (bh * SS + (size_t)kb * 64) * KDIM;
            const bf16* gv = Vt + bh * KDIM * SS + (size_t)kb * 64;
#pragma unroll
            for (int i = 0; i < 4; i++) {
                int c = t + i * 128;
                int r = c >> 3, off = (c & 7) * 8;
                stage16(&Kt[r * AT_LD + off], gk + (size_t)r * KDIM + off);
                stage16(&Vs[r * AT_LD + off], gv + (size_t)r * SS + off);
            }
        }
        async_wait0();
        __syncthreads();

        // S = Q @ K^T : 4 column tiles x 2 K-steps
        v8f sacc[4] = {};
#pragma unroll
        for (int ct = 0; ct < 4; ct++) {
            sacc[ct] = wmma_bf16(qf0, load_b_frag(&Kt[(ct * 16) * AT_LD],      AT_LD), sacc[ct]);
            sacc[ct] = wmma_bf16(qf1, load_b_frag(&Kt[(ct * 16) * AT_LD + 32], AT_LD), sacc[ct]);
        }

        // Online softmax. Row r+8*half lives across the 16 lanes of this half:
        // xor-shuffle masks 1,2,4,8 reduce within the aligned 16-lane group.
        float pv[4][8];
#pragma unroll
        for (int r = 0; r < 8; r++) {
            float mx = fmaxf(fmaxf(sacc[0][r], sacc[1][r]),
                             fmaxf(sacc[2][r], sacc[3][r]));
#pragma unroll
            for (int sh = 1; sh <= 8; sh <<= 1) mx = fmaxf(mx, __shfl_xor(mx, sh, 32));
            float mnew = fmaxf(mrow[r], mx * scale);
            float sum = 0.0f;
#pragma unroll
            for (int ct = 0; ct < 4; ct++) {
                float p = __expf(sacc[ct][r] * scale - mnew);
                pv[ct][r] = p;
                sum += p;
            }
#pragma unroll
            for (int sh = 1; sh <= 8; sh <<= 1) sum += __shfl_xor(sum, sh, 32);
            float alpha = __expf(mrow[r] - mnew);
            lrow[r] = lrow[r] * alpha + sum;
            mrow[r] = mnew;
#pragma unroll
            for (int kdt = 0; kdt < 4; kdt++) oacc[kdt][r] *= alpha;
        }

        // Stage P (bf16) in this wave's private LDS region; same-wave RAW only,
        // so a dscnt wait suffices (no workgroup barrier).
#pragma unroll
        for (int ct = 0; ct < 4; ct++)
#pragma unroll
            for (int r = 0; r < 8; r++)
                Pw[(r + 8 * half) * AT_LD + ct * 16 + nl] = (bf16)pv[ct][r];
        asm volatile("s_wait_dscnt 0" ::: "memory");

        // O += P @ V : P re-read as A-fragments, V^T rows are B-fragments.
        v16bf pa0 = load_a_frag(Pw,      AT_LD);
        v16bf pa1 = load_a_frag(Pw + 32, AT_LD);
#pragma unroll
        for (int kdt = 0; kdt < 4; kdt++) {
            oacc[kdt] = wmma_bf16(pa0, load_b_frag(&Vs[(kdt * 16) * AT_LD],      AT_LD), oacc[kdt]);
            oacc[kdt] = wmma_bf16(pa1, load_b_frag(&Vs[(kdt * 16) * AT_LD + 32], AT_LD), oacc[kdt]);
        }
    }

    // Normalize and emit attended as [b][s][h*KD+kd] bf16.
#pragma unroll
    for (int kdt = 0; kdt < 4; kdt++) {
#pragma unroll
        for (int r = 0; r < 8; r++) {
            int s   = qb * 64 + wave * 16 + r + 8 * half;
            int col = h * KDIM + kdt * 16 + nl;
            float val = oacc[kdt][r] / lrow[r];
            attw[(size_t)(b * SS + s) * DD + col] = (bf16)val;
        }
    }
}

// ---------------------------------------------------------------------------
extern "C" void kernel_launch(void* const* d_in, const int* in_sizes, int n_in,
                              void* d_out, int out_size, void* d_ws, size_t ws_size,
                              hipStream_t stream) {
    (void)in_sizes; (void)n_in; (void)out_size; (void)ws_size;
    const float* x  = (const float*)d_in[0];
    const float* Wq = (const float*)d_in[1];
    const float* bq = (const float*)d_in[2];
    const float* Wk = (const float*)d_in[3];
    const float* bk = (const float*)d_in[4];
    const float* Wv = (const float*)d_in[5];
    const float* bv = (const float*)d_in[6];
    const float* Wo = (const float*)d_in[7];
    const float* bo = (const float*)d_in[8];

    const size_t NX = (size_t)BB * SS * DD;  // 4M elements
    const size_t NW = (size_t)DD * DD;       // 1M elements

    char* ws = (char*)d_ws;
    bf16* xb   = (bf16*)ws; ws += NX * sizeof(bf16);
    bf16* wqb  = (bf16*)ws; ws += NW * sizeof(bf16);  // W^T bf16
    bf16* wkb  = (bf16*)ws; ws += NW * sizeof(bf16);
    bf16* wvb  = (bf16*)ws; ws += NW * sizeof(bf16);
    bf16* wob  = (bf16*)ws; ws += NW * sizeof(bf16);
    bf16* qw   = (bf16*)ws; ws += NX * sizeof(bf16);
    bf16* kw   = (bf16*)ws; ws += NX * sizeof(bf16);
    bf16* vw   = (bf16*)ws; ws += NX * sizeof(bf16);
    bf16* attw = (bf16*)ws; ws += NX * sizeof(bf16);

    // fp32 -> bf16 convert (x) and transpose-convert (weights)
    cvt_kernel<<<(int)((NX / 4 + 255) / 256), 256, 0, stream>>>((const float4*)x, (v4bf*)xb, (int)(NX / 4));
    dim3 tg(DD / 32, DD / 32);
    cvt_t_kernel<<<tg, 256, 0, stream>>>(Wq, wqb, DD, DD);
    cvt_t_kernel<<<tg, 256, 0, stream>>>(Wk, wkb, DD, DD);
    cvt_t_kernel<<<tg, 256, 0, stream>>>(Wv, wvb, DD, DD);
    cvt_t_kernel<<<tg, 256, 0, stream>>>(Wo, wob, DD, DD);

    // QKV projections (Q/K row-major per head, V pre-transposed per head)
    dim3 gg(BB * SS / GBM, DD / GBN);
    gemm_kernel<<<gg, 256, 0, stream>>>(xb, wqb, bq, qw, BB * SS, DD, DD, 0);
    gemm_kernel<<<gg, 256, 0, stream>>>(xb, wkb, bk, kw, BB * SS, DD, DD, 0);
    gemm_kernel<<<gg, 256, 0, stream>>>(xb, wvb, bv, vw, BB * SS, DD, DD, 1);

    // Fused flash attention
    attn_kernel<<<BB * HH * (SS / 64), 128, 0, stream>>>(qw, kw, vw, attw);

    // Output projection -> fp32 d_out
    gemm_kernel<<<gg, 256, 0, stream>>>(attw, wob, bo, d_out, BB * SS, DD, DD, 2);
}
